// TrAttentionBase_11836929867974
// MI455X (gfx1250) — compile-verified, matrix operand order checked by reading, not run on hardware
//
#include <hip/hip_runtime.h>
#include <math.h>

// ---------------------------------------------------------------------------
// TrAttention (transposed channel attention) for MI455X / gfx1250, wave32.
// B=2 T=4 N=2 DIM=64 H=W=128 HEADS=4  ->  IMG=16 images of (64,128,128) fp32.
// Bandwidth-bound (~1.5 GB HBM traffic vs ~12 GFLOP): keep fp32 in HBM,
// convert to f16 only inside v_wmma_f32_16x16x32_f16 fragments (f32 accum).
// ---------------------------------------------------------------------------

typedef __attribute__((ext_vector_type(16))) _Float16 v16h;
typedef __attribute__((ext_vector_type(8)))  float    v8f;

#define DIMC   64
#define HWSZ   16384      // 128*128
#define IMGN   16         // B*T*N
#define BHN    32         // (B*T)*HEADS
#define NROWS  32         // N*C attention rows per (bt,head)

__device__ __forceinline__ float4 ld4(const float* p) {
    return *(const float4*)p;
}

// Pack 16 fp32 (four float4 runs) into a v16h WMMA fragment.
__device__ __forceinline__ v16h pack16(float4 a, float4 b, float4 c, float4 d) {
    v16h h;
    h[0]=(_Float16)a.x;  h[1]=(_Float16)a.y;  h[2]=(_Float16)a.z;  h[3]=(_Float16)a.w;
    h[4]=(_Float16)b.x;  h[5]=(_Float16)b.y;  h[6]=(_Float16)b.z;  h[7]=(_Float16)b.w;
    h[8]=(_Float16)c.x;  h[9]=(_Float16)c.y;  h[10]=(_Float16)c.z; h[11]=(_Float16)c.w;
    h[12]=(_Float16)d.x; h[13]=(_Float16)d.y; h[14]=(_Float16)d.z; h[15]=(_Float16)d.w;
    return h;
}

// attention row r (0..31) of (bt,head)  ->  flat (img*64+ch) row index
__device__ __forceinline__ int attn_row_ch(int bt, int head, int r) {
    return (bt * 2 + (r >> 4)) * DIMC + head * 16 + (r & 15);
}

// ---------------------------------------------------------------------------
// K1/K6: 1x1 conv = 64x64 GEMM over channels, N = spatial (16384 per image).
// Block = 128 threads (4 waves). LDS-staged 64x64 fp32 activation tile
// (row pad 65 to avoid half-wave bank conflicts when building B fragments).
// Each wave: one 16-col N-tile, sweeps 4 M-tiles x 2 K-steps = 8 WMMAs.
// ---------------------------------------------------------------------------
__global__ __launch_bounds__(128)
void conv1x1_wmma_k(const float* __restrict__ x, const float* __restrict__ w,
                    const float* __restrict__ bias, float* __restrict__ y) {
    __shared__ float xs[64][65];
    const int img = blockIdx.y;
    const int sp0 = blockIdx.x * 64;
    const int tid = threadIdx.x;
    const size_t ibase = (size_t)img * DIMC * HWSZ;

    // cooperative coalesced float4 load of the 64(ch) x 64(sp) tile
    for (int idx = tid; idx < 1024; idx += 128) {
        const int ch = idx >> 4, seg = idx & 15;
        const float4 v = ld4(x + ibase + (size_t)ch * HWSZ + sp0 + seg * 4);
        xs[ch][seg*4+0] = v.x; xs[ch][seg*4+1] = v.y;
        xs[ch][seg*4+2] = v.z; xs[ch][seg*4+3] = v.w;
    }
    __syncthreads();

    const int wv = tid >> 5, lane = tid & 31;
    const int ln = lane & 15;
    const bool hi = lane >= 16;
    const int kb = hi ? 8 : 0;          // lane-half K base (ISA 16-bit layout)
    const int col = wv * 16 + ln;       // this wave's spatial column

    // B fragments (activations) for both K-steps, reused across all M-tiles
    v16h bfr[2];
    #pragma unroll
    for (int ks = 0; ks < 2; ++ks) {
        #pragma unroll
        for (int j = 0; j < 8; ++j) {
            bfr[ks][j]     = (_Float16)xs[ks*32 + kb + j][col];
            bfr[ks][8 + j] = (_Float16)xs[ks*32 + kb + 16 + j][col];
        }
    }

    #pragma unroll
    for (int mt = 0; mt < 4; ++mt) {
        v8f acc;
        #pragma unroll
        for (int r = 0; r < 8; ++r)          // fuse bias into the accumulator
            acc[r] = bias[mt*16 + r + (hi ? 8 : 0)];
        #pragma unroll
        for (int ks = 0; ks < 2; ++ks) {
            const float* wr = w + (mt*16 + ln) * DIMC + ks*32 + kb;  // 16B aligned
            const v16h af = pack16(ld4(wr), ld4(wr+4), ld4(wr+16), ld4(wr+20));
            acc = __builtin_amdgcn_wmma_f32_16x16x32_f16(
                false, af, false, bfr[ks], (short)0, acc, false, false);
        }
        #pragma unroll
        for (int r = 0; r < 8; ++r) {
            const int row = mt*16 + r + (hi ? 8 : 0);
            y[ibase + (size_t)row * HWSZ + sp0 + col] = acc[r];
        }
    }
}

// ---------------------------------------------------------------------------
// K2: depthwise 3x3, SAME padding (zeros). One (img,ch) 32x32 tile per block,
// halo tile staged in LDS. Pure VALU/memory kernel.
// ---------------------------------------------------------------------------
__global__ __launch_bounds__(256)
void dwconv3x3_k(const float* __restrict__ x, const float* __restrict__ wd,
                 const float* __restrict__ bd, float* __restrict__ y) {
    __shared__ float tl[34][36];
    const int tile = blockIdx.x, ch = blockIdx.y, img = blockIdx.z;
    const int tx0 = (tile & 3) * 32, ty0 = (tile >> 2) * 32;
    const size_t base = ((size_t)img * DIMC + ch) * HWSZ;
    const int tid = threadIdx.y * 32 + threadIdx.x;

    for (int idx = tid; idx < 34 * 34; idx += 256) {
        const int ly = idx / 34, lx = idx % 34;
        const int gy = ty0 + ly - 1, gx = tx0 + lx - 1;
        float v = 0.f;
        if ((unsigned)gy < 128u && (unsigned)gx < 128u) v = x[base + gy * 128 + gx];
        tl[ly][lx] = v;
    }
    __syncthreads();

    float wk[9];
    #pragma unroll
    for (int i = 0; i < 9; ++i) wk[i] = wd[ch * 9 + i];
    const float bb = bd[ch];
    const int tx = threadIdx.x;

    for (int ry = threadIdx.y; ry < 32; ry += 8) {
        float s = bb;
        #pragma unroll
        for (int ki = 0; ki < 3; ++ki)
            #pragma unroll
            for (int kj = 0; kj < 3; ++kj)
                s += wk[ki * 3 + kj] * tl[ry + ki][tx + kj];
        y[base + (size_t)(ty0 + ry) * 128 + tx0 + tx] = s;
    }
}

// ---------------------------------------------------------------------------
// K3: L2 row norm over the 16384-long spatial axis, one (img,ch) per block.
// ---------------------------------------------------------------------------
__global__ __launch_bounds__(256)
void rownorm_k(const float* __restrict__ x, float* __restrict__ nrm) {
    __shared__ float red[256];
    const int row = blockIdx.x, tid = threadIdx.x;
    const float4* xv = (const float4*)(x + (size_t)row * HWSZ);
    float s = 0.f;
    #pragma unroll 4
    for (int j = 0; j < 16; ++j) {
        const float4 v = xv[tid + j * 256];
        s += v.x*v.x + v.y*v.y + v.z*v.z + v.w*v.w;
    }
    red[tid] = s; __syncthreads();
    for (int off = 128; off; off >>= 1) {
        if (tid < off) red[tid] += red[tid + off];
        __syncthreads();
    }
    if (tid == 0) nrm[row] = sqrtf(red[0]);
}

// ---------------------------------------------------------------------------
// K4: Gram matrix S = q . k^T over K=16384 + temperature/norm scaling +
// softmax, fused. One block per (bt,head); 4 waves = 2x2 output tiles;
// 512 WMMA K-steps each. l2norm applied post-hoc as row/col 1/(|q||k|).
// ---------------------------------------------------------------------------
__global__ __launch_bounds__(128)
void gram_softmax_k(const float* __restrict__ qf, const float* __restrict__ kf,
                    const float* __restrict__ qn, const float* __restrict__ kn,
                    const float* __restrict__ temp, float* __restrict__ attnP) {
    __shared__ float S[32][33];
    const int bh = blockIdx.x, bt = bh >> 2, head = bh & 3;
    const int tid = threadIdx.x, wv = tid >> 5, lane = tid & 31;
    const int mt = wv & 1, nt = wv >> 1;
    const int ln = lane & 15;
    const bool hi = lane >= 16;
    const int kb = hi ? 8 : 0;

    const float* qrow = qf + ((size_t)attn_row_ch(bt, head, mt*16 + ln) << 14);
    const float* krow = kf + ((size_t)attn_row_ch(bt, head, nt*16 + ln) << 14);

    v8f acc = {};
    #pragma unroll 2
    for (int d0 = 0; d0 < HWSZ; d0 += 32) {
        const float* qa = qrow + d0 + kb;        // all 16B aligned runs
        const float* ka = krow + d0 + kb;
        const v16h af = pack16(ld4(qa), ld4(qa+4), ld4(qa+16), ld4(qa+20));
        const v16h bf = pack16(ld4(ka), ld4(ka+4), ld4(ka+16), ld4(ka+20));
        acc = __builtin_amdgcn_wmma_f32_16x16x32_f16(
            false, af, false, bf, (short)0, acc, false, false);
    }

    const float ts = temp[head];
    #pragma unroll
    for (int r = 0; r < 8; ++r) {
        const int grow = mt*16 + r + (hi ? 8 : 0);
        const int gcol = nt*16 + ln;
        const float nq = fmaxf(qn[attn_row_ch(bt, head, grow)], 1e-12f);
        const float nk = fmaxf(kn[attn_row_ch(bt, head, gcol)], 1e-12f);
        S[grow][gcol] = acc[r] * ts / (nq * nk);
    }
    __syncthreads();

    if (tid < 32) {                               // softmax over 32 cols
        float mx = -3.4e38f;
        #pragma unroll
        for (int j = 0; j < 32; ++j) mx = fmaxf(mx, S[tid][j]);
        float e[32], sum = 0.f;
        #pragma unroll
        for (int j = 0; j < 32; ++j) { e[j] = __expf(S[tid][j] - mx); sum += e[j]; }
        const float inv = 1.f / sum;
        #pragma unroll
        for (int j = 0; j < 32; ++j) attnP[bh * 1024 + tid * 32 + j] = e[j] * inv;
    }
}

// ---------------------------------------------------------------------------
// K5: out = attn(32x32) @ v(32x16384). LDS-staged 32x64 v tile per block;
// 8 waves = 2 M-tiles x 4 N-tiles, one K=32 WMMA each.
// ---------------------------------------------------------------------------
__global__ __launch_bounds__(256)
void av_wmma_k(const float* __restrict__ attnP, const float* __restrict__ vf,
               float* __restrict__ ao) {
    __shared__ float vs[32][65];
    const int bh = blockIdx.y, bt = bh >> 2, head = bh & 3;
    const int sp0 = blockIdx.x * 64;
    const int tid = threadIdx.x;

    for (int idx = tid; idx < 512; idx += 256) {
        const int row = idx >> 4, seg = idx & 15;
        const float4 v = ld4(vf + ((size_t)attn_row_ch(bt, head, row) << 14)
                             + sp0 + seg * 4);
        vs[row][seg*4+0] = v.x; vs[row][seg*4+1] = v.y;
        vs[row][seg*4+2] = v.z; vs[row][seg*4+3] = v.w;
    }
    __syncthreads();

    const int wv = tid >> 5, lane = tid & 31;
    const int mt = wv & 1, nt = wv >> 1;          // nt in 0..3
    const int ln = lane & 15;
    const bool hi = lane >= 16;
    const int kb = hi ? 8 : 0;
    const int col = nt * 16 + ln;

    const float* ar = attnP + bh * 1024 + (mt*16 + ln) * 32 + kb;
    const v16h af = pack16(ld4(ar), ld4(ar+4), ld4(ar+16), ld4(ar+20));

    v16h bf;
    #pragma unroll
    for (int j = 0; j < 8; ++j) {
        bf[j]     = (_Float16)vs[kb + j][col];
        bf[8 + j] = (_Float16)vs[kb + 16 + j][col];
    }

    v8f acc = {};
    acc = __builtin_amdgcn_wmma_f32_16x16x32_f16(
        false, af, false, bf, (short)0, acc, false, false);

    #pragma unroll
    for (int r = 0; r < 8; ++r) {
        const int grow = mt*16 + r + (hi ? 8 : 0);
        ao[((size_t)attn_row_ch(bt, head, grow) << 14) + sp0 + col] = acc[r];
    }
}

// ---------------------------------------------------------------------------
extern "C" void kernel_launch(void* const* d_in, const int* in_sizes, int n_in,
                              void* d_out, int out_size, void* d_ws, size_t ws_size,
                              hipStream_t stream) {
    (void)in_sizes; (void)n_in; (void)out_size; (void)ws_size;
    const float* q    = (const float*)d_in[0];
    const float* k    = (const float*)d_in[1];
    const float* v    = (const float*)d_in[2];
    const float* w1q  = (const float*)d_in[3];
    const float* b1q  = (const float*)d_in[4];
    const float* w1k  = (const float*)d_in[5];
    const float* b1k  = (const float*)d_in[6];
    const float* w1v  = (const float*)d_in[7];
    const float* b1v  = (const float*)d_in[8];
    const float* wdq  = (const float*)d_in[9];
    const float* bdq  = (const float*)d_in[10];
    const float* wdk  = (const float*)d_in[11];
    const float* bdk  = (const float*)d_in[12];
    const float* wdv  = (const float*)d_in[13];
    const float* bdv  = (const float*)d_in[14];
    const float* wout = (const float*)d_in[15];
    const float* bout = (const float*)d_in[16];
    const float* temp = (const float*)d_in[17];

    float* ws = (float*)d_ws;
    const size_t IMGSZ = (size_t)IMGN * DIMC * HWSZ;   // 16.8M floats each
    float* tmp   = ws;                   // conv1x1 staging; reused as attn-out
    float* qf    = ws + 1 * IMGSZ;
    float* kf    = ws + 2 * IMGSZ;
    float* vf    = ws + 3 * IMGSZ;
    float* qn    = ws + 4 * IMGSZ;       // 1024 floats
    float* kn    = qn + 1024;            // 1024 floats
    float* attnP = kn + 1024;            // 32*32*32 floats

    const dim3 cgrid(HWSZ / 64, IMGN), cblk(128);
    const dim3 dgrid(16, DIMC, IMGN), dblk(32, 8);

    conv1x1_wmma_k<<<cgrid, cblk, 0, stream>>>(q, w1q, b1q, tmp);
    dwconv3x3_k  <<<dgrid, dblk, 0, stream>>>(tmp, wdq, bdq, qf);
    conv1x1_wmma_k<<<cgrid, cblk, 0, stream>>>(k, w1k, b1k, tmp);
    dwconv3x3_k  <<<dgrid, dblk, 0, stream>>>(tmp, wdk, bdk, kf);
    conv1x1_wmma_k<<<cgrid, cblk, 0, stream>>>(v, w1v, b1v, tmp);
    dwconv3x3_k  <<<dgrid, dblk, 0, stream>>>(tmp, wdv, bdv, vf);

    rownorm_k<<<IMGN * DIMC, 256, 0, stream>>>(qf, qn);
    rownorm_k<<<IMGN * DIMC, 256, 0, stream>>>(kf, kn);

    gram_softmax_k<<<BHN, 128, 0, stream>>>(qf, kf, qn, kn, temp, attnP);

    av_wmma_k<<<dim3(HWSZ / 64, BHN), 256, 0, stream>>>(attnP, vf, tmp);

    conv1x1_wmma_k<<<cgrid, cblk, 0, stream>>>(tmp, wout, bout, (float*)d_out);
}
